// BasicBlock_31894427140848
// MI455X (gfx1250) — matrix operand order repaired and checked by reading, not compiled, not run on hardware
//
#include <hip/hip_runtime.h>

// ---------------------------------------------------------------------------
// MI455X (gfx1250) implementation of one MDCP/RDN dehazing iteration.
// Conv layers (~166 GFLOP) run as implicit-GEMM on v_wmma_f32_16x16x32_bf16
// with a channel-innermost K order so every fragment is built from b128 loads
// with immediate offsets. Elementwise/reduction/dark-channel are f32 kernels.
// ---------------------------------------------------------------------------

typedef __attribute__((ext_vector_type(16))) __bf16 v16bf;
typedef __attribute__((ext_vector_type(8)))  __bf16 v8bf;
typedef __attribute__((ext_vector_type(8)))  float  v8f;

constexpr int W_   = 384;
constexpr int H_   = 384;
constexpr int HW_  = W_ * H_;          // 147456
constexpr int S1_  = 4 * HW_;          // 589824
constexpr int S3_  = 12 * HW_;         // 1769472
constexpr int NHW_ = S1_;              // pixels per channel plane (batch-flat)

constexpr float LAM1 = 1.0f, LAM2 = 0.01f, LAM3 = 0.001f, LAM4 = 0.01f, LAM5 = 0.01f;

// d_out layout (reference tuple order, flattened)
constexpr size_t OFF_B = 0;
constexpr size_t OFF_T = (size_t)S3_;
constexpr size_t OFF_J = 2 * (size_t)S3_;
constexpr size_t OFF_G = 3 * (size_t)S3_;
constexpr size_t OFF_H = 4 * (size_t)S3_;
constexpr size_t OFF_P = OFF_H + S1_;
constexpr size_t OFF_Q = OFF_P + S1_;
constexpr size_t OFF_U = OFF_Q + S1_;
constexpr size_t OFF_V = OFF_U + S3_;
constexpr size_t OFF_X = OFF_V + S3_;
constexpr size_t OFF_Y = OFF_X + S3_;
constexpr size_t OFF_GAM = OFF_Y + S3_;

// ---------------------------------------------------------------------------
// Weight conversion: f32 OIHW -> bf16 [co][Kpad], K order = (ky*KW+kx)*CPAD+ci
// (channels innermost). co padded to 32, ci padded to CPAD with zeros.
// ---------------------------------------------------------------------------
__global__ __launch_bounds__(256)
void wconv_kernel(const float* __restrict__ src, __bf16* __restrict__ dst,
                  int Cout, int Cin, int KK, int CPAD, int KPAD)
{
    int i = blockIdx.x * 256 + threadIdx.x;
    int tot = 32 * KPAD;
    if (i >= tot) return;
    int co = i / KPAD;
    int k  = i - co * KPAD;
    int q  = k / CPAD;           // ky*KW+kx
    int ci = k - q * CPAD;
    float v = (co < Cout && ci < Cin) ? src[(size_t)(co * Cin + ci) * KK + q] : 0.f;
    dst[(size_t)co * KPAD + k] = (__bf16)v;
}

// ---------------------------------------------------------------------------
// Background light update: per-(b,c) spatial mean of the elementwise solve.
// ---------------------------------------------------------------------------
__global__ __launch_bounds__(256)
void bmean_kernel(const float* __restrict__ I, const float* __restrict__ t,
                  const float* __restrict__ J, const float* __restrict__ Bp,
                  float* __restrict__ Bmean)
{
    int g = blockIdx.x;                 // b*3+c in [0,12)
    size_t base = (size_t)g * HW_;
    float s = 0.f;
    for (int i = threadIdx.x; i < HW_; i += 256) {
        float tv = t[base + i], Jv = J[base + i];
        float Iv = I[base + i], Bv = Bp[base + i];
        float omt = 1.f - tv;
        float num = LAM3 * Bv - LAM1 * (Jv * tv - Iv) * omt;
        float den = LAM1 * omt * omt + LAM3;
        s += num / den;
    }
    __shared__ float red[256];
    red[threadIdx.x] = s;
    __syncthreads();
    for (int off = 128; off; off >>= 1) {
        if (threadIdx.x < off) red[threadIdx.x] += red[threadIdx.x + off];
        __syncthreads();
    }
    if (threadIdx.x == 0) Bmean[g] = red[0] * (1.f / (float)HW_);
}

// ---------------------------------------------------------------------------
// t update + 1x1 (3->1) conv + broadcast; emits RDN input actIn = t + X/g4.
// ---------------------------------------------------------------------------
__global__ __launch_bounds__(256)
void stage2_kernel(const float* __restrict__ tp, const float* __restrict__ Hin,
                   const float* __restrict__ Iin, const float* __restrict__ Jin,
                   const float* __restrict__ Xin, const float* __restrict__ Bmean,
                   const float* __restrict__ t1d, const float* __restrict__ gamma,
                   float* __restrict__ tout, float* __restrict__ tbuf,
                   __bf16* __restrict__ actIn)
{
    int p = blockIdx.x * 256 + threadIdx.x;
    if (p >= S1_) return;
    int b = p / HW_;
    int yx = p - b * HW_;
    float g4 = gamma[3];
    float Hv = Hin[p];
    float t1 = 0.f;
    float xs[3];
    #pragma unroll
    for (int c = 0; c < 3; ++c) {
        size_t idx = (size_t)(b * 3 + c) * HW_ + yx;
        float Bm = Bmean[b * 3 + c];
        float Iv = Iin[idx], Jv = Jin[idx], Xv = Xin[idx], tpv = tp[idx];
        float JB = Jv - Bm;
        float tc = (LAM2 * tpv + g4 * Hv - LAM1 * (Bm - Iv) * JB - Xv)
                 / (LAM1 * JB * JB + LAM2 + g4);
        t1 += t1d[c] * tc;
        xs[c] = Xv;
    }
    tbuf[p] = t1;
    #pragma unroll
    for (int c = 0; c < 3; ++c) {
        size_t idx = (size_t)(b * 3 + c) * HW_ + yx;
        tout[idx] = t1;
        actIn[(size_t)c * NHW_ + p] = (__bf16)(t1 + xs[c] / g4);
    }
}

// ---------------------------------------------------------------------------
// J, u, v, Y updates + B broadcast + G passthrough.
// ---------------------------------------------------------------------------
__global__ __launch_bounds__(256)
void stage3_kernel(const float* __restrict__ Iin, const float* __restrict__ Gin,
                   const float* __restrict__ uin, const float* __restrict__ vin,
                   const float* __restrict__ Yin, const float* __restrict__ Bmean,
                   const float* __restrict__ tbuf, const float* __restrict__ gamma,
                   float* __restrict__ Bout, float* __restrict__ Jout,
                   float* __restrict__ Gout, float* __restrict__ uout,
                   float* __restrict__ vout, float* __restrict__ Yout)
{
    int i = blockIdx.x * 256 + threadIdx.x;
    if (i >= S3_) return;
    int bc = i / HW_;
    int b = bc / 3;
    int yx = i - bc * HW_;
    float g1 = gamma[0], g2 = gamma[1], g3 = gamma[2], g4 = gamma[3], g5 = gamma[4];
    float t1 = tbuf[b * HW_ + yx];
    float Bm = Bmean[bc];
    float Iv = Iin[i], Gv = Gin[i], uo = uin[i], vo = vin[i], Yo = Yin[i];
    float den = LAM1 * t1 * t1 + g3 + g4 + g5;
    float Jn = (LAM1 * (t1 * (Iv - Bm * (1.f - t1))) + g3 * Gv + g4 * uo - g5 * vo - Yo + g5) / den;
    float un = (g1 * uo + g4 * Jn) / (g1 + g4);
    float vn = (g2 * vo - g5 * Jn + g5) / (g2 + g5);
    float Yn = Yo + g3 * (Jn - Gv);
    Bout[i] = Bm; Jout[i] = Jn; Gout[i] = Gv;
    uout[i] = un; vout[i] = vn; Yout[i] = Yn;
}

// ---------------------------------------------------------------------------
// Implicit-GEMM conv on WMMA (bf16 in / f32 acc).
//
//   K order: k = (ky*3+kx)*CPAD + ci  (CPAD = Cin padded to mult of 32).
//   LDS tile (3x3): [4 rows][66 cols][CPAD ch], channels innermost, so each
//   8-consecutive-K run of the A layout is one 16B ds_load_b128.
//   Weights: [co][KPAD] same K order -> B fragment = two global b128 loads.
//
//   Block = 128 threads (4 waves), covers 2 rows x 64 cols = 128 pixels.
//   Wave = 32 px x 32 ch = 2x2 WMMA tiles (A and B fragments each reused 2x).
//   K loop fully unrolled: all LDS/global accesses use immediate offsets.
//
//   A frag (16-bit 16x32): lane L -> M=L%16, K = (L>=16?8:0)+(i<8?i:i+8).
//   B frag (32x16):        lane L -> N=L%16, K = (L>=16?8:0)+(i<8?i:i+8).
//   D:  VGPR r, lane L ->  M = r + (L>=16?8:0), N = L%16.
// ---------------------------------------------------------------------------
template<int KSZ, int CPAD>
__global__ __launch_bounds__(128)
void conv_wmma(const __bf16* __restrict__ in, const __bf16* __restrict__ wgt,
               const float* __restrict__ bias,
               __bf16* __restrict__ out, float* __restrict__ outf,
               const __bf16* __restrict__ resid, __bf16* __restrict__ out2,
               int Cin, int CoutValid, int relu)
{
    constexpr int KPAD   = (KSZ == 3) ? 9 * CPAD : CPAD;
    constexpr int KSTEPS = KPAD / 32;
    constexpr int COLS   = (KSZ == 3) ? 66 : 128;
    constexpr int ROWS   = (KSZ == 3) ? 4 : 1;
    __shared__ __bf16 smem[ROWS * COLS * CPAD];

    const int tid = threadIdx.x;
    int b = 0, y = 0, x0 = 0, p0 = 0;
    if (KSZ == 3) {
        int bid = blockIdx.x;            // 4 * 192 * 6 blocks
        b = bid / 1152;
        int r = bid - b * 1152;
        int yp = r / 6;
        int xt = r - yp * 6;
        y = yp * 2;
        x0 = xt * 64;
    } else {
        p0 = blockIdx.x * 128;
    }

    // ---- stage activation tile to LDS (channels innermost) ----
    if (KSZ == 3) {
        for (int i = tid; i < CPAD * 264; i += 128) {
            int ci = i / 264;
            int rc = i - ci * 264;
            int r4 = rc / 66;
            int col = rc - r4 * 66;
            int yy = y + r4 - 1;
            int xx = x0 + col - 1;
            __bf16 v = (__bf16)0.f;      // 'SAME' zero pad + channel pad
            if (ci < Cin && yy >= 0 && yy < H_ && xx >= 0 && xx < W_)
                v = in[(size_t)ci * NHW_ + b * HW_ + yy * W_ + xx];
            smem[(r4 * 66 + col) * CPAD + ci] = v;
        }
    } else {
        for (int i = tid; i < CPAD * 128; i += 128) {
            int ci = i >> 7;
            int col = i & 127;
            smem[col * CPAD + ci] = in[(size_t)ci * NHW_ + p0 + col];
        }
    }
    __syncthreads();

    const int lane = tid & 31;
    const int wv   = tid >> 5;
    const int m16  = lane & 15;
    const int hi   = lane >> 4;

    int wy = 0, wx = 0;
    if (KSZ == 3) { wy = wv >> 1; wx = wv & 1; }

    // per-lane A base (element index into smem), one per pixel sub-tile
    int aBase[2];
    #pragma unroll
    for (int pt = 0; pt < 2; ++pt) {
        int mloc = (KSZ == 3) ? (wx * 32 + pt * 16 + m16)
                              : (wv * 32 + pt * 16 + m16);
        int rowc = (KSZ == 3) ? (wy * 66 + mloc) : mloc;
        aBase[pt] = rowc * CPAD + hi * 8;
    }
    // per-lane B base pointers, one per channel sub-tile
    const __bf16* wB[2];
    #pragma unroll
    for (int ct = 0; ct < 2; ++ct)
        wB[ct] = wgt + (size_t)(ct * 16 + m16) * KPAD + hi * 8;

    v8f acc[2][2] = {};   // [ct][pt]

    #pragma unroll
    for (int s = 0; s < KSTEPS; ++s) {
        const int k0 = s * 32;
        const int q  = k0 / CPAD;            // constant-folded
        const int cb = k0 - q * CPAD;
        int cA;
        if (KSZ == 3) {
            const int ky = q / 3;
            const int kx = q - ky * 3;
            cA = (ky * 66 + kx) * CPAD + cb;
        } else {
            cA = cb;
        }
        v16bf a[2], bf[2];
        #pragma unroll
        for (int pt = 0; pt < 2; ++pt) {
            const __bf16* pA = &smem[aBase[pt] + cA];
            v8bf lo = *(const v8bf*)pA;          // K run k0+hi*8 .. +7
            v8bf hi8 = *(const v8bf*)(pA + 16);  // K run k0+hi*8+16 .. +23
            a[pt] = __builtin_shufflevector(lo, hi8,
                        0,1,2,3,4,5,6,7,8,9,10,11,12,13,14,15);
        }
        #pragma unroll
        for (int ct = 0; ct < 2; ++ct) {
            const __bf16* pB = wB[ct] + k0;
            v8bf lo = *(const v8bf*)pB;
            v8bf hi8 = *(const v8bf*)(pB + 16);
            bf[ct] = __builtin_shufflevector(lo, hi8,
                        0,1,2,3,4,5,6,7,8,9,10,11,12,13,14,15);
        }
        #pragma unroll
        for (int ct = 0; ct < 2; ++ct)
            #pragma unroll
            for (int pt = 0; pt < 2; ++pt)
                acc[ct][pt] = __builtin_amdgcn_wmma_f32_16x16x32_bf16(
                    false, a[pt], false, bf[ct], (short)0, acc[ct][pt], false, false);
    }

    // ---- epilogue: bias / residual / relu / stores ----
    #pragma unroll
    for (int ct = 0; ct < 2; ++ct) {
        int co = ct * 16 + m16;
        if (co >= CoutValid) continue;
        float bv = bias ? bias[co] : 0.f;
        #pragma unroll
        for (int pt = 0; pt < 2; ++pt) {
            #pragma unroll
            for (int r = 0; r < 8; ++r) {
                int pix;
                if (KSZ == 3)
                    pix = b * HW_ + (y + wy) * W_ + x0 + wx * 32 + pt * 16 + hi * 8 + r;
                else
                    pix = p0 + wv * 32 + pt * 16 + hi * 8 + r;
                float v = acc[ct][pt][r] + bv;
                if (resid) v += (float)resid[(size_t)co * NHW_ + pix];
                if (relu)  v = v > 0.f ? v : 0.f;
                if (out)   out [(size_t)co * NHW_ + pix] = (__bf16)v;
                if (out2)  out2[(size_t)co * NHW_ + pix] = (__bf16)v;
                if (outf && co == 0) outf[pix] = v;
            }
        }
    }
}

// ---------------------------------------------------------------------------
// H output + X dual update (+ gamma[2] scalar output).
// ---------------------------------------------------------------------------
__global__ __launch_bounds__(256)
void xh_kernel(const float* __restrict__ Xin, const float* __restrict__ Hf,
               const float* __restrict__ tbuf, const float* __restrict__ gamma,
               float* __restrict__ Hout, float* __restrict__ Xout,
               float* __restrict__ gamOut)
{
    int p = blockIdx.x * 256 + threadIdx.x;
    if (p >= S1_) return;
    float g4 = gamma[3];
    float h = Hf[p];
    Hout[p] = h;
    float t1 = tbuf[p];
    int b = p / HW_;
    int yx = p - b * HW_;
    #pragma unroll
    for (int c = 0; c < 3; ++c) {
        size_t i = (size_t)(b * 3 + c) * HW_ + yx;
        Xout[i] = Xin[i] + g4 * (t1 - h);
    }
    if (p == 0) gamOut[0] = gamma[2];
}

// ---------------------------------------------------------------------------
// Dark channel, separable 7x7 reflect-padded min + soft threshold.
// ---------------------------------------------------------------------------
__device__ __forceinline__ int reflect_idx(int x, int n)
{
    if (x < 0) return -x;
    if (x >= n) return 2 * n - 2 - x;
    return x;
}

__global__ __launch_bounds__(256)
void darkH_kernel(const float* __restrict__ src, float* __restrict__ tmp)
{
    int p = blockIdx.x * 256 + threadIdx.x;
    if (p >= S1_) return;
    int b = p / HW_;
    int yx = p - b * HW_;
    int y = yx / W_;
    int x = yx - y * W_;
    float m = 3.0e38f;
    #pragma unroll
    for (int dx = -3; dx <= 3; ++dx) {
        int xr = reflect_idx(x + dx, W_);
        size_t base = (size_t)(b * 3) * HW_ + y * W_ + xr;
        float cm = fminf(src[base], fminf(src[base + HW_], src[base + 2 * HW_]));
        m = fminf(m, cm);
    }
    tmp[p] = m;
}

__global__ __launch_bounds__(256)
void darkV_kernel(const float* __restrict__ tmp, float* __restrict__ dst,
                  const float* __restrict__ gamma, float lamNum)
{
    int p = blockIdx.x * 256 + threadIdx.x;
    if (p >= S1_) return;
    int b = p / HW_;
    int yx = p - b * HW_;
    int y = yx / W_;
    int x = yx - y * W_;
    float m = 3.0e38f;
    #pragma unroll
    for (int dy = -3; dy <= 3; ++dy) {
        int yr = reflect_idx(y + dy, H_);
        m = fminf(m, tmp[b * HW_ + yr * W_ + x]);
    }
    float lam = lamNum / gamma[0];
    float a = fabsf(m) - lam;
    a = a > 0.f ? a : 0.f;
    dst[p] = copysignf(a, m);      // sign(m)*relu(|m|-lam)
}

// ---------------------------------------------------------------------------
// Host-side orchestration (graph-capture safe: launches only).
// ---------------------------------------------------------------------------
struct WLayer { int wi, bi, Cin, KK, Cout; };
static const WLayer g_wls[13] = {
    {16, 17,   3, 9, 32},   // 0  sfe1
    {18, 19,  32, 9, 32},   // 1  sfe2
    {20, 21,  32, 9, 32},   // 2  b0_l0
    {22, 23,  64, 9, 32},   // 3  b0_l1
    {24, 25,  96, 9, 32},   // 4  b0_l2
    {26, 27, 128, 1, 32},   // 5  b0_lff
    {28, 29,  32, 9, 32},   // 6  b1_l0
    {30, 31,  64, 9, 32},   // 7  b1_l1
    {32, 33,  96, 9, 32},   // 8  b1_l2
    {34, 35, 128, 1, 32},   // 9  b1_lff
    {36, 37,  64, 1, 32},   // 10 gff1
    {38, 39,  32, 9, 32},   // 11 gff2
    {40, 41,  32, 9,  1},   // 12 out
};
static inline int cpad_of(int Cin) { return Cin < 32 ? 32 : Cin; }
static inline int kpad_of(const WLayer& w) { return w.KK == 9 ? 9 * cpad_of(w.Cin) : cpad_of(w.Cin); }

extern "C" void kernel_launch(void* const* d_in, const int* in_sizes, int n_in,
                              void* d_out, int out_size, void* d_ws, size_t ws_size,
                              hipStream_t stream)
{
    (void)in_sizes; (void)n_in; (void)out_size; (void)ws_size;

    const float* I    = (const float*)d_in[0];
    const float* t_p  = (const float*)d_in[1];
    const float* B_p  = (const float*)d_in[2];
    const float* t_o  = (const float*)d_in[4];
    const float* J_o  = (const float*)d_in[5];
    const float* G    = (const float*)d_in[6];
    const float* u_o  = (const float*)d_in[7];
    const float* v_o  = (const float*)d_in[8];
    const float* X_o  = (const float*)d_in[9];
    const float* Y_o  = (const float*)d_in[10];
    const float* H_o  = (const float*)d_in[11];
    const float* gam  = (const float*)d_in[14];
    const float* t1d  = (const float*)d_in[15];
    float* out = (float*)d_out;

    // ---- workspace carving ----
    char* ws = (char*)d_ws;
    size_t off = 0;
    auto alloc = [&](size_t bytes) -> char* {
        char* p = ws + off;
        off = (off + bytes + 255) & ~(size_t)255;
        return p;
    };

    size_t wo[14]; wo[0] = 0;
    for (int l = 0; l < 13; ++l) wo[l + 1] = wo[l] + (size_t)32 * kpad_of(g_wls[l]);

    __bf16* wbAll = (__bf16*)alloc(wo[13] * sizeof(__bf16));
    float*  Bmean = (float*)alloc(12 * sizeof(float));
    float*  tbuf  = (float*)alloc((size_t)S1_ * sizeof(float));
    float*  Hf32  = (float*)alloc((size_t)S1_ * sizeof(float));
    float*  tmpd  = (float*)alloc((size_t)S1_ * sizeof(float));
    __bf16* actIn = (__bf16*)alloc((size_t)3   * NHW_ * sizeof(__bf16));
    __bf16* f1b   = (__bf16*)alloc((size_t)32  * NHW_ * sizeof(__bf16));
    __bf16* hb    = (__bf16*)alloc((size_t)128 * NHW_ * sizeof(__bf16));
    __bf16* fe    = (__bf16*)alloc((size_t)64  * NHW_ * sizeof(__bf16));
    __bf16* gb    = (__bf16*)alloc((size_t)32  * NHW_ * sizeof(__bf16));

    // ---- weight conversion ----
    for (int l = 0; l < 13; ++l) {
        const WLayer& wl = g_wls[l];
        int Kp = kpad_of(wl), Cp = cpad_of(wl.Cin);
        int tot = 32 * Kp;
        wconv_kernel<<<(tot + 255) / 256, 256, 0, stream>>>(
            (const float*)d_in[wl.wi], wbAll + wo[l], wl.Cout, wl.Cin, wl.KK, Cp, Kp);
    }

    // ---- elementwise pipeline ----
    bmean_kernel<<<12, 256, 0, stream>>>(I, t_o, J_o, B_p, Bmean);
    stage2_kernel<<<(S1_ + 255) / 256, 256, 0, stream>>>(
        t_p, H_o, I, J_o, X_o, Bmean, t1d, gam, out + OFF_T, tbuf, actIn);
    stage3_kernel<<<(S3_ + 255) / 256, 256, 0, stream>>>(
        I, G, u_o, v_o, Y_o, Bmean, tbuf, gam,
        out + OFF_B, out + OFF_J, out + OFF_G, out + OFF_U, out + OFF_V, out + OFF_Y);

    // ---- RDN on WMMA ----
    const dim3 cg(4608);                 // 3x3: 4*192*6 blocks; 1x1: NHW/128
    auto bias = [&](int l) { return (const float*)d_in[g_wls[l].bi]; };
    __bf16* nb = nullptr;
    float*  nf = nullptr;

    conv_wmma<3, 32><<<cg, 128, 0, stream>>>(actIn, wbAll + wo[0], bias(0), f1b, nf, nb, nb, 3, 32, 0);
    conv_wmma<3, 32><<<cg, 128, 0, stream>>>(f1b,   wbAll + wo[1], bias(1), hb,  nf, nb, nb, 32, 32, 0);

    for (int bi2 = 0; bi2 < 2; ++bi2) {
        int l0 = 2 + bi2 * 4;
        conv_wmma<3, 32><<<cg, 128, 0, stream>>>(
            hb, wbAll + wo[l0 + 0], bias(l0 + 0), hb + (size_t)32 * NHW_, nf, nb, nb, 32, 32, 1);
        conv_wmma<3, 64><<<cg, 128, 0, stream>>>(
            hb, wbAll + wo[l0 + 1], bias(l0 + 1), hb + (size_t)64 * NHW_, nf, nb, nb, 64, 32, 1);
        conv_wmma<3, 96><<<cg, 128, 0, stream>>>(
            hb, wbAll + wo[l0 + 2], bias(l0 + 2), hb + (size_t)96 * NHW_, nf, nb, nb, 96, 32, 1);
        __bf16* dup = (bi2 == 0) ? hb : nb;   // block0 LFF result seeds block1's base
        conv_wmma<1, 128><<<cg, 128, 0, stream>>>(
            hb, wbAll + wo[l0 + 3], bias(l0 + 3), fe + (size_t)bi2 * 32 * NHW_,
            nf, hb, dup, 128, 32, 0);
    }

    conv_wmma<1, 64><<<cg, 128, 0, stream>>>(fe, wbAll + wo[10], bias(10), gb, nf, nb, nb, 64, 32, 0);
    conv_wmma<3, 32><<<cg, 128, 0, stream>>>(gb, wbAll + wo[11], bias(11), hb, nf, f1b, nb, 32, 32, 0);
    conv_wmma<3, 32><<<cg, 128, 0, stream>>>(hb, wbAll + wo[12], bias(12), nb, Hf32, nb, nb, 32, 1, 0);

    // ---- H/X outputs, gamma scalar ----
    xh_kernel<<<(S1_ + 255) / 256, 256, 0, stream>>>(
        X_o, Hf32, tbuf, gam, out + OFF_H, out + OFF_X, out + OFF_GAM);

    // ---- dark channel + soft threshold ----
    darkH_kernel<<<(S1_ + 255) / 256, 256, 0, stream>>>(out + OFF_U, tmpd);
    darkV_kernel<<<(S1_ + 255) / 256, 256, 0, stream>>>(tmpd, out + OFF_P, gam, LAM4);
    darkH_kernel<<<(S1_ + 255) / 256, 256, 0, stream>>>(out + OFF_V, tmpd);
    darkV_kernel<<<(S1_ + 255) / 256, 256, 0, stream>>>(tmpd, out + OFF_Q, gam, LAM5);
}
